// SparseLinear_68015102099869
// MI455X (gfx1250) — matrix operand order repaired and checked by reading, not compiled, not run on hardware
//
#include <hip/hip_runtime.h>
#include <stdint.h>

// ---------------------------------------------------------------------------
// SparseLinear forward as dense bf16 WMMA GEMM (MI455X / gfx1250, wave32).
// out[b,o] = sum_k x[b,k] * W[o,k]; bias is dead in the reference.
// M=8192 (batch), N=4096 (out), K=4096 (in).
// Pass 1: fp32 -> bf16 into d_ws (96MB; fits the 192MB L2).
// Pass 2: 128x128x32 tiled GEMM, 8 wave32 per block, each wave 4x2 tiles of
//         v_wmma_f32_16x16x32_bf16. Staging uses GLOBAL_LOAD_ASYNC_TO_LDS
//         (ASYNCcnt) when the toolchain exposes it, double-buffered in LDS.
// ---------------------------------------------------------------------------

typedef __attribute__((ext_vector_type(16))) __bf16 v16bf;
typedef __attribute__((ext_vector_type(8)))  __bf16 v8bf;
typedef __attribute__((ext_vector_type(8)))  float  v8f;
typedef int b128v __attribute__((__vector_size__(16)));  // builtin's pointee

#define M_DIM 8192
#define N_DIM 4096
#define K_DIM 4096
#define BM 128
#define BN 128
#define BK 32
#define LDA 40   // BK + 8 bf16 pad: 80B row stride, 16B aligned, spreads banks

#define AS1 __attribute__((address_space(1)))
#define AS3 __attribute__((address_space(3)))

#if __has_builtin(__builtin_amdgcn_global_load_async_to_lds_b128)
#define HAVE_ASYNC_LDS 1
#else
#define HAVE_ASYNC_LDS 0
#endif

__device__ __forceinline__ void wait_async0() {
#if __has_builtin(__builtin_amdgcn_s_wait_asynccnt)
  __builtin_amdgcn_s_wait_asynccnt(0);
#else
  asm volatile("s_wait_asynccnt 0x0" ::: "memory");
#endif
}

__device__ __forceinline__ uint32_t pack2_bf16(float lo, float hi) {
  union { float f; uint32_t u; } a, b;
  a.f = lo; b.f = hi;
  uint32_t x = (a.u + 0x7fffu + ((a.u >> 16) & 1u)) >> 16;  // RNE truncate
  uint32_t y = (b.u + 0x7fffu + ((b.u >> 16) & 1u)) >> 16;
  return (y << 16) | (x & 0xffffu);
}

// fp32 -> bf16 conversion, 8 floats per thread (two 16B loads -> one 16B store)
__global__ __launch_bounds__(256) void cvt_f32_bf16_kernel(
    const float* __restrict__ src, uint4* __restrict__ dst, int n_vec) {
  int i = blockIdx.x * 256 + threadIdx.x;
  if (i >= n_vec) return;
  const float4* s = (const float4*)src + (size_t)i * 2;
  float4 f0 = s[0];
  float4 f1 = s[1];
  uint4 o;
  o.x = pack2_bf16(f0.x, f0.y);
  o.y = pack2_bf16(f0.z, f0.w);
  o.z = pack2_bf16(f1.x, f1.y);
  o.w = pack2_bf16(f1.z, f1.w);
  dst[i] = o;
}

template <bool SRC_BF16>
__global__ __launch_bounds__(256) void gemm_bf16_wmma_kernel(
    const void* __restrict__ Aptr, const void* __restrict__ Bptr,
    float* __restrict__ C) {
  __shared__ __bf16 As[2][BM * LDA];
  __shared__ __bf16 Bs[2][BN * LDA];

  const int tid  = threadIdx.x;
  const int lane = tid & 31;
  const int wave = tid >> 5;
  const int wm   = (wave & 1) * 64;   // wave M offset in block tile
  const int wn   = (wave >> 1) * 32;  // wave N offset in block tile

  const int bm = blockIdx.y * BM;
  const int bn = blockIdx.x * BN;

  // staging: 2 threads per 32-elem row; each thread: two 8-elem (16B) chunks
  const int srow = tid >> 1;          // 0..127
  const int scol = (tid & 1) * 16;    // 0 or 16

  const __bf16* Ab = (const __bf16*)Aptr;
  const __bf16* Bb = (const __bf16*)Bptr;
  const float*  Af = (const float*)Aptr;
  const float*  Bf = (const float*)Bptr;

  v8f acc[4][2];
#pragma unroll
  for (int mi = 0; mi < 4; ++mi)
#pragma unroll
    for (int ni = 0; ni < 2; ++ni) acc[mi][ni] = (v8f)0.0f;

  // ISA fragment addressing (16-bit A 16x32; B 32x16; wave32):
  //   A: lane<16 -> row M=lane, K {0..7,16..23}; lane>=16 -> K {8..15,24..31}
  //   B: lane<16 -> col N=lane, K {0..7,16..23}; lane>=16 -> K {8..15,24..31}
  const int fm  = lane & 15;
  const int fh  = lane >> 4;
  const int ac0 = fh * 8;
  const int ac1 = 16 + fh * 8;
  const int bc0 = fh * 16;

  auto compute = [&](int buf) {
    const __bf16* Ap = As[buf];
    const __bf16* Bp = Bs[buf];
    v16bf afr[4];
#pragma unroll
    for (int mi = 0; mi < 4; ++mi) {
      const __bf16* p = Ap + (wm + mi * 16 + fm) * LDA;
      v8bf lo = *(const v8bf*)(p + ac0);
      v8bf hi = *(const v8bf*)(p + ac1);
      afr[mi] = __builtin_shufflevector(lo, hi, 0, 1, 2, 3, 4, 5, 6, 7, 8, 9,
                                        10, 11, 12, 13, 14, 15);
    }
    v16bf bfr[2];
#pragma unroll
    for (int ni = 0; ni < 2; ++ni) {
      const __bf16* p = Bp + (wn + ni * 16 + fm) * LDA;
      v8bf lo = *(const v8bf*)(p + ac0);
      v8bf hi = *(const v8bf*)(p + ac1);
      bfr[ni] = __builtin_shufflevector(lo, hi, 0, 1, 2, 3, 4, 5, 6, 7, 8, 9,
                                        10, 11, 12, 13, 14, 15);
    }
#pragma unroll
    for (int mi = 0; mi < 4; ++mi)
#pragma unroll
      for (int ni = 0; ni < 2; ++ni)
        acc[mi][ni] = __builtin_amdgcn_wmma_f32_16x16x32_bf16(
            false, afr[mi], false, bfr[ni], (short)0, acc[mi][ni], false,
            false);
  };

#if HAVE_ASYNC_LDS
  if constexpr (SRC_BF16) {
    // -------- async global->LDS staging path (ASYNCcnt, no VGPR bounce) ----
    // IOFFSET applies to BOTH the global and LDS addresses (ISA 08 §4.4), so
    // the second 16B chunk reuses the same pointers with offset=16.
    auto issue = [&](int buf, int k0) {
      const __bf16* ga = Ab + (size_t)(bm + srow) * K_DIM + k0 + scol;
      const __bf16* gb = Bb + (size_t)(bn + srow) * K_DIM + k0 + scol;
      __bf16* la = &As[buf][srow * LDA + scol];
      __bf16* lb = &Bs[buf][srow * LDA + scol];
      __builtin_amdgcn_global_load_async_to_lds_b128(
          (AS1 b128v*)ga, (AS3 b128v*)la, 0, 0);
      __builtin_amdgcn_global_load_async_to_lds_b128(
          (AS1 b128v*)ga, (AS3 b128v*)la, 16, 0);
      __builtin_amdgcn_global_load_async_to_lds_b128(
          (AS1 b128v*)gb, (AS3 b128v*)lb, 0, 0);
      __builtin_amdgcn_global_load_async_to_lds_b128(
          (AS1 b128v*)gb, (AS3 b128v*)lb, 16, 0);
    };
    issue(0, 0);
    wait_async0();
    __syncthreads();
    int buf = 0;
    for (int k0 = BK; k0 < K_DIM; k0 += BK) {
      issue(buf ^ 1, k0);  // DMA next tile while computing current
      compute(buf);
      wait_async0();
      __syncthreads();
      buf ^= 1;
    }
    compute(buf);
  } else
#endif
  {
    // -------- register-staged path (also handles fp32 inline convert) ------
    uint4 ra0, ra1, rb0, rb1;
    auto stage_global = [&](int k0) {
      if constexpr (SRC_BF16) {
        const uint4* pa =
            (const uint4*)(Ab + (size_t)(bm + srow) * K_DIM + k0 + scol);
        ra0 = pa[0]; ra1 = pa[1];
        const uint4* pb =
            (const uint4*)(Bb + (size_t)(bn + srow) * K_DIM + k0 + scol);
        rb0 = pb[0]; rb1 = pb[1];
      } else {
        const float4* pa =
            (const float4*)(Af + (size_t)(bm + srow) * K_DIM + k0 + scol);
        float4 a0 = pa[0], a1 = pa[1], a2 = pa[2], a3 = pa[3];
        ra0.x = pack2_bf16(a0.x, a0.y); ra0.y = pack2_bf16(a0.z, a0.w);
        ra0.z = pack2_bf16(a1.x, a1.y); ra0.w = pack2_bf16(a1.z, a1.w);
        ra1.x = pack2_bf16(a2.x, a2.y); ra1.y = pack2_bf16(a2.z, a2.w);
        ra1.z = pack2_bf16(a3.x, a3.y); ra1.w = pack2_bf16(a3.z, a3.w);
        const float4* pb =
            (const float4*)(Bf + (size_t)(bn + srow) * K_DIM + k0 + scol);
        float4 b0 = pb[0], b1 = pb[1], b2 = pb[2], b3 = pb[3];
        rb0.x = pack2_bf16(b0.x, b0.y); rb0.y = pack2_bf16(b0.z, b0.w);
        rb0.z = pack2_bf16(b1.x, b1.y); rb0.w = pack2_bf16(b1.z, b1.w);
        rb1.x = pack2_bf16(b2.x, b2.y); rb1.y = pack2_bf16(b2.z, b2.w);
        rb1.z = pack2_bf16(b3.x, b3.y); rb1.w = pack2_bf16(b3.z, b3.w);
      }
    };
    auto stage_lds = [&](int buf) {
      *(uint4*)&As[buf][srow * LDA + scol]     = ra0;
      *(uint4*)&As[buf][srow * LDA + scol + 8] = ra1;
      *(uint4*)&Bs[buf][srow * LDA + scol]     = rb0;
      *(uint4*)&Bs[buf][srow * LDA + scol + 8] = rb1;
    };

    stage_global(0);
    stage_lds(0);
    __syncthreads();
    int buf = 0;
    for (int k0 = BK; k0 < K_DIM; k0 += BK) {
      stage_global(k0);   // prefetch next tile into registers
      compute(buf);       // consume current LDS tile
      stage_lds(buf ^ 1); // write the other buffer: no extra barrier needed
      __syncthreads();
      buf ^= 1;
    }
    compute(buf);
  }

  // C/D layout: VGPR r -> (M = r + 8*(lane>=16), N = lane&15)
#pragma unroll
  for (int mi = 0; mi < 4; ++mi) {
#pragma unroll
    for (int ni = 0; ni < 2; ++ni) {
      const int row0 = bm + wm + mi * 16 + fh * 8;
      const int col  = bn + wn + ni * 16 + fm;
#pragma unroll
      for (int r = 0; r < 8; ++r)
        C[(size_t)(row0 + r) * N_DIM + col] = acc[mi][ni][r];
    }
  }
}

extern "C" void kernel_launch(void* const* d_in, const int* in_sizes, int n_in,
                              void* d_out, int out_size, void* d_ws,
                              size_t ws_size, hipStream_t stream) {
  const float* x = (const float*)d_in[0];  // [8192, 4096]
  const float* W = (const float*)d_in[1];  // [4096, 4096]
  // d_in[2] = bias: dead in the reference (overwritten before use)
  float* out = (float*)d_out;              // [8192, 4096]

  const size_t bytesA = (size_t)M_DIM * K_DIM * 2;  // x as bf16: 64MB
  const size_t bytesB = (size_t)N_DIM * K_DIM * 2;  // W as bf16: 32MB

  dim3 grid(N_DIM / BN, M_DIM / BM);  // 32 x 64
  dim3 block(256);

  if (ws_size >= bytesA + bytesB) {
    uint4* xb = (uint4*)d_ws;
    uint4* wb = (uint4*)((char*)d_ws + bytesA);
    const int nvx = (M_DIM * (long)K_DIM) / 8;  // 4,194,304
    const int nvw = (N_DIM * (long)K_DIM) / 8;  // 2,097,152
    cvt_f32_bf16_kernel<<<(nvx + 255) / 256, 256, 0, stream>>>(x, xb, nvx);
    cvt_f32_bf16_kernel<<<(nvw + 255) / 256, 256, 0, stream>>>(W, wb, nvw);
    gemm_bf16_wmma_kernel<true><<<grid, block, 0, stream>>>(xb, wb, out);
  } else {
    // Workspace too small: convert fp32->bf16 inline while staging to LDS.
    gemm_bf16_wmma_kernel<false><<<grid, block, 0, stream>>>(x, W, out);
  }
}
